// StackedRNNLayer_82566451298755
// MI455X (gfx1250) — matrix operand order repaired and checked by reading, not compiled, not run on hardware
//
#include <hip/hip_runtime.h>

// ---------------------------------------------------------------------------
// Types for CDNA5 WMMA (wave32, 16x16x32 bf16 -> f32 accumulate)
// ---------------------------------------------------------------------------
typedef __attribute__((ext_vector_type(16))) __bf16          v16bf;
typedef __attribute__((ext_vector_type(16))) unsigned short  v16u;
typedef __attribute__((ext_vector_type(8)))  unsigned short  v8u;
typedef __attribute__((ext_vector_type(8)))  float           v8f;
typedef __attribute__((ext_vector_type(4)))  float           v4f;
typedef __attribute__((ext_vector_type(4)))  unsigned short  v4u;

__device__ __forceinline__ unsigned short f32_to_bf16_rne(float f) {
  unsigned int u = __builtin_bit_cast(unsigned int, f);
  unsigned int r = u + 0x7FFFu + ((u >> 16) & 1u);   // round-to-nearest-even
  return (unsigned short)(r >> 16);
}

// LDS address (low 32 bits of flat shared pointer) for async-to-LDS dst.
__device__ __forceinline__ unsigned lds_addr32(const void* p) {
  return (unsigned)(unsigned long long)p;
}

// CDNA5 async copy: global -> LDS, 16 bytes per lane, tracked by ASYNCcnt.
__device__ __forceinline__ void async_g2l_b128(unsigned lds, const void* g) {
  asm volatile("global_load_async_to_lds_b128 %0, %1, off"
               :: "v"(lds), "v"((unsigned long long)g)
               : "memory");
}
// Wait until at most N async copies are still outstanding (in-order complete).
template <int N>
__device__ __forceinline__ void wait_async_le() {
  asm volatile("s_wait_asynccnt %0" :: "n"(N) : "memory");
}

// A-matrix frag (16x32 bf16 tile, row-major in LDS, stride `stride` elems).
// Lane l holds row M = l%16; element j -> K = (j/8)*16 + half*8 + (j%8).
__device__ __forceinline__ v16bf load_a_frag(const unsigned short* sA, int row,
                                             int stride, int half) {
  const v8u lo = *(const v8u*)&sA[row * stride + half * 8];
  const v8u hi = *(const v8u*)&sA[row * stride + 16 + half * 8];
  v16u t;
#pragma unroll
  for (int i = 0; i < 8; ++i) { t[i] = lo[i]; t[i + 8] = hi[i]; }
  return __builtin_bit_cast(v16bf, t);
}

// B-matrix frag (32x16 bf16 tile). LDS holds B as [N][K]; lane l holds col
// N = l%16; element j -> K = (j/8)*16 + half*8 + (j%8).
__device__ __forceinline__ v16bf load_b_frag(const unsigned short* sBt, int col,
                                             int stride, int half) {
  const v8u lo = *(const v8u*)&sBt[col * stride + half * 8];
  const v8u hi = *(const v8u*)&sBt[col * stride + 16 + half * 8];
  v16u t;
#pragma unroll
  for (int i = 0; i < 8; ++i) { t[i] = lo[i]; t[i + 8] = hi[i]; }
  return __builtin_bit_cast(v16bf, t);
}

__device__ __forceinline__ v8f wmma_bf16(v16bf a, v16bf b, v8f c) {
  return __builtin_amdgcn_wmma_f32_16x16x32_bf16(false, a, false, b,
                                                 (short)0, c, false, false);
}

// ---------------------------------------------------------------------------
// Pre-pass kernels.
// ---------------------------------------------------------------------------
__global__ void cvt_bf16_vec4(const float* __restrict__ in,
                              unsigned short* __restrict__ out, int n4) {
  int i = blockIdx.x * blockDim.x + threadIdx.x;
  if (i < n4) {
    v4f f = ((const v4f*)in)[i];
    v4u o;
#pragma unroll
    for (int j = 0; j < 4; ++j) o[j] = f32_to_bf16_rne(f[j]);
    ((v4u*)out)[i] = o;
  }
}

__global__ __launch_bounds__(256) void transpose_cvt_bf16(
    const float* __restrict__ in,        // [K][N] fp32
    unsigned short* __restrict__ out,    // [N][K] bf16
    int K, int N) {
  __shared__ float tile[32][33];
  const int n0 = blockIdx.x * 32, k0 = blockIdx.y * 32;
  const int tx = threadIdx.x & 31, ty = threadIdx.x >> 5;  // 32 x 8
#pragma unroll
  for (int i = 0; i < 4; ++i) {
    int k = ty + i * 8;
    tile[k][tx] = in[(size_t)(k0 + k) * N + (n0 + tx)];
  }
  __syncthreads();
#pragma unroll
  for (int i = 0; i < 4; ++i) {
    int n = ty + i * 8;
    out[(size_t)(n0 + n) * K + (k0 + tx)] = f32_to_bf16_rne(tile[tx][n]);
  }
}

__global__ void zero_u32(unsigned int* __restrict__ p, int n) {
  int i = blockIdx.x * blockDim.x + threadIdx.x;
  if (i < n) p[i] = 0u;
}

// ---------------------------------------------------------------------------
// Big GEMM: C[M,N] = A[M,K] * Bt[N,K]^T (+ bias). A, Bt bf16 in memory.
// Double-buffered GLOBAL_LOAD_ASYNC_TO_LDS_B128 staging pipelined against
// WMMA. Last K-tile is peeled so the steady-state loop is branch-free:
// issue 3 asyncs -> s_wait_asynccnt 3 -> barrier -> 4 WMMA -> barrier.
// Block tile 128x64, 256 threads = 8 waves; wave = 32x32 (2x2 WMMA tiles).
// ---------------------------------------------------------------------------
#define GBM 128
#define GBN 64
#define GBK 32

__global__ __launch_bounds__(256) void gemm_bf16_async(
    const unsigned short* __restrict__ A,    // [M][K] bf16
    const unsigned short* __restrict__ Bt,   // [N][K] bf16 (pre-transposed)
    const float* __restrict__ bias,          // [N] or null
    float* __restrict__ C,                   // [M][N] fp32 or null
    unsigned short* __restrict__ Cbf,        // [M][N] bf16 or null
    int M, int N, int K) {
  __shared__ unsigned short sA[2][GBM * GBK];   // [m][k]
  __shared__ unsigned short sB[2][GBN * GBK];   // [n][k]

  const int tid  = threadIdx.x;
  const int lane = tid & 31;
  const int wave = tid >> 5;       // 0..7
  const int wm   = wave >> 1;      // 0..3 : 32 rows each
  const int wn   = wave & 1;       // 0..1 : 32 cols each
  const int m0   = blockIdx.y * GBM;
  const int n0   = blockIdx.x * GBN;
  const int half = lane >> 4;
  const int l16  = lane & 15;

  // staging coordinates (fixed per thread)
  const int ar0 = tid >> 2, ac0 = (tid & 3) * 8;          // A chunk 0
  const int ar1 = (tid + 256) >> 2, ac1 = ac0;            // A chunk 1
  const int br  = tid >> 2, bc = (tid & 3) * 8;           // B chunk

  // per-thread source pointers (advance by GBK each tile)
  const unsigned short* pa0 = &A[(size_t)(m0 + ar0) * K + ac0];
  const unsigned short* pa1 = &A[(size_t)(m0 + ar1) * K + ac1];
  const unsigned short* pb  = &Bt[(size_t)(n0 + br) * K + bc];

  v8f acc00 = {}, acc01 = {}, acc10 = {}, acc11 = {};

  const int NT = K / GBK;
  // prologue: issue tile 0 into buffer 0
  async_g2l_b128(lds_addr32(&sA[0][ar0 * GBK + ac0]), pa0);
  async_g2l_b128(lds_addr32(&sA[0][ar1 * GBK + ac1]), pa1);
  async_g2l_b128(lds_addr32(&sB[0][br * GBK + bc]),   pb);

  for (int i = 0; i < NT - 1; ++i) {
    const int buf = i & 1, nb = buf ^ 1;
    pa0 += GBK; pa1 += GBK; pb += GBK;
    async_g2l_b128(lds_addr32(&sA[nb][ar0 * GBK + ac0]), pa0);
    async_g2l_b128(lds_addr32(&sA[nb][ar1 * GBK + ac1]), pa1);
    async_g2l_b128(lds_addr32(&sB[nb][br * GBK + bc]),   pb);
    wait_async_le<3>();   // tile i landed; tile i+1 still in flight
    __syncthreads();

    v16bf a0 = load_a_frag(sA[buf], wm * 32 + l16,      GBK, half);
    v16bf a1 = load_a_frag(sA[buf], wm * 32 + 16 + l16, GBK, half);
    v16bf b0 = load_b_frag(sB[buf], wn * 32 + l16,      GBK, half);
    v16bf b1 = load_b_frag(sB[buf], wn * 32 + 16 + l16, GBK, half);

    acc00 = wmma_bf16(a0, b0, acc00);
    acc01 = wmma_bf16(a0, b1, acc01);
    acc10 = wmma_bf16(a1, b0, acc10);
    acc11 = wmma_bf16(a1, b1, acc11);
    __syncthreads();   // safe to overwrite buf on next iteration's issue
  }
  {  // peeled last tile
    const int buf = (NT - 1) & 1;
    wait_async_le<0>();
    __syncthreads();
    v16bf a0 = load_a_frag(sA[buf], wm * 32 + l16,      GBK, half);
    v16bf a1 = load_a_frag(sA[buf], wm * 32 + 16 + l16, GBK, half);
    v16bf b0 = load_b_frag(sB[buf], wn * 32 + l16,      GBK, half);
    v16bf b1 = load_b_frag(sB[buf], wn * 32 + 16 + l16, GBK, half);
    acc00 = wmma_bf16(a0, b0, acc00);
    acc01 = wmma_bf16(a0, b1, acc01);
    acc10 = wmma_bf16(a1, b0, acc10);
    acc11 = wmma_bf16(a1, b1, acc11);
  }

  // Epilogue: C/D layout -> M = v + 8*half, N = l16
  const float bv0 = bias ? bias[n0 + wn * 32 + l16]      : 0.f;
  const float bv1 = bias ? bias[n0 + wn * 32 + 16 + l16] : 0.f;
  const int r0b = m0 + wm * 32 + 8 * half;
  const int c0b = n0 + wn * 32 + l16;
  if (C) {
#pragma unroll
    for (int v = 0; v < 8; ++v) {
      C[(size_t)(r0b + v) * N + c0b]           = acc00[v] + bv0;
      C[(size_t)(r0b + v) * N + c0b + 16]      = acc01[v] + bv1;
      C[(size_t)(r0b + v + 16) * N + c0b]      = acc10[v] + bv0;
      C[(size_t)(r0b + v + 16) * N + c0b + 16] = acc11[v] + bv1;
    }
  }
  if (Cbf) {
#pragma unroll
    for (int v = 0; v < 8; ++v) {
      Cbf[(size_t)(r0b + v) * N + c0b]           = f32_to_bf16_rne(acc00[v] + bv0);
      Cbf[(size_t)(r0b + v) * N + c0b + 16]      = f32_to_bf16_rne(acc01[v] + bv1);
      Cbf[(size_t)(r0b + v + 16) * N + c0b]      = f32_to_bf16_rne(acc10[v] + bv0);
      Cbf[(size_t)(r0b + v + 16) * N + c0b + 16] = f32_to_bf16_rne(acc11[v] + bv1);
    }
  }
}

// ---------------------------------------------------------------------------
// Fused GRU recurrent step. Strip = 64 batch rows x 64 H-columns x 3 gates.
// Grid = H/64 = 16 blocks, 256 threads (8 waves). Per K-tile each thread
// issues exactly 4 async copies (256 A-chunks + 768 B-chunks = 1024), so the
// double-buffer wait threshold is uniform (4). Each wave: 1 A-frag reused by
// 3 gates x 2 N-tiles = 6 WMMAs per K-tile. Last tile peeled.
// ---------------------------------------------------------------------------
#define BSZ 64
#define HD  1024
#define G3  3072
#define SCT 64
#define SKT 32

__global__ __launch_bounds__(256) void gru_step(
    const float* __restrict__ h_prev,            // [64][1024] fp32
    const unsigned short* __restrict__ h_prev_bf,// [64][1024] bf16
    float* __restrict__ h_out,                   // [64][1024] fp32
    unsigned short* __restrict__ h_out_bf,       // [64][1024] bf16
    const float* __restrict__ gx_t,              // [64][3072]  (r|z|n)
    const float* __restrict__ pad_t,             // [64]
    const unsigned short* __restrict__ Whh_t) {  // [3072][1024] bf16
  __shared__ unsigned short sA[2][BSZ * SKT];        // h tile [m][k]
  __shared__ unsigned short sB[2][3][SCT * SKT];     // per-gate Whh tile [n][k]

  const int tid  = threadIdx.x;
  const int lane = tid & 31;
  const int wave = tid >> 5;
  const int wm   = wave >> 1;     // 0..3 : rows wm*16
  const int wp   = wave & 1;      // 0..1 : col pair wp*32 + {0,16}
  const int half = lane >> 4;
  const int l16  = lane & 15;
  const int c0   = blockIdx.x * SCT;

  // staging coordinates: A chunk (1/thread), B chunks (3/thread, one per gate)
  const int amr = tid >> 2, akc = (tid & 3) * 8;
  const int bnr = tid >> 2, bkc = akc;

  const unsigned short* pa = &h_prev_bf[(size_t)amr * HD + akc];
  const unsigned short* pb0 = &Whh_t[(size_t)(0 * HD + c0 + bnr) * HD + bkc];
  const unsigned short* pb1 = &Whh_t[(size_t)(1 * HD + c0 + bnr) * HD + bkc];
  const unsigned short* pb2 = &Whh_t[(size_t)(2 * HD + c0 + bnr) * HD + bkc];

  v8f zacc = {};
  v8f acc[3][2];
#pragma unroll
  for (int g = 0; g < 3; ++g) { acc[g][0] = zacc; acc[g][1] = zacc; }

  const int NT = HD / SKT;
  // prologue: tile 0 -> buffer 0
  async_g2l_b128(lds_addr32(&sA[0][amr * SKT + akc]),    pa);
  async_g2l_b128(lds_addr32(&sB[0][0][bnr * SKT + bkc]), pb0);
  async_g2l_b128(lds_addr32(&sB[0][1][bnr * SKT + bkc]), pb1);
  async_g2l_b128(lds_addr32(&sB[0][2][bnr * SKT + bkc]), pb2);

  for (int i = 0; i < NT - 1; ++i) {
    const int buf = i & 1, nb = buf ^ 1;
    pa += SKT; pb0 += SKT; pb1 += SKT; pb2 += SKT;
    async_g2l_b128(lds_addr32(&sA[nb][amr * SKT + akc]),    pa);
    async_g2l_b128(lds_addr32(&sB[nb][0][bnr * SKT + bkc]), pb0);
    async_g2l_b128(lds_addr32(&sB[nb][1][bnr * SKT + bkc]), pb1);
    async_g2l_b128(lds_addr32(&sB[nb][2][bnr * SKT + bkc]), pb2);
    wait_async_le<4>();
    __syncthreads();

    v16bf a = load_a_frag(sA[buf], wm * 16 + l16, SKT, half);
#pragma unroll
    for (int g = 0; g < 3; ++g) {
#pragma unroll
      for (int j = 0; j < 2; ++j) {
        v16bf b = load_b_frag(sB[buf][g], wp * 32 + j * 16 + l16, SKT, half);
        acc[g][j] = wmma_bf16(a, b, acc[g][j]);
      }
    }
    __syncthreads();
  }
  {  // peeled last tile
    const int buf = (NT - 1) & 1;
    wait_async_le<0>();
    __syncthreads();
    v16bf a = load_a_frag(sA[buf], wm * 16 + l16, SKT, half);
#pragma unroll
    for (int g = 0; g < 3; ++g) {
#pragma unroll
      for (int j = 0; j < 2; ++j) {
        v16bf b = load_b_frag(sB[buf][g], wp * 32 + j * 16 + l16, SKT, half);
        acc[g][j] = wmma_bf16(a, b, acc[g][j]);
      }
    }
  }

  // Gate fusion. D layout: M = v + 8*half, N = l16.
#pragma unroll
  for (int j = 0; j < 2; ++j) {
#pragma unroll
    for (int v = 0; v < 8; ++v) {
      int row = wm * 16 + v + 8 * half;
      int col = c0 + wp * 32 + j * 16 + l16;
      float gxr = gx_t[row * G3 + col];
      float gxz = gx_t[row * G3 + HD + col];
      float gxn = gx_t[row * G3 + 2 * HD + col];
      float r = 1.f / (1.f + __expf(-(gxr + acc[0][j][v])));
      float z = 1.f / (1.f + __expf(-(gxz + acc[1][j][v])));
      float n = tanhf(gxn + r * acc[2][j][v]);
      float hp = h_prev[row * HD + col];
      float hn = (1.f - z) * n + z * hp;
      float p  = pad_t[row];
      hn = p * hp + (1.f - p) * hn;
      h_out[row * HD + col] = hn;
      h_out_bf[row * HD + col] = f32_to_bf16_rne(hn);
    }
  }
}

// ---------------------------------------------------------------------------
// Host orchestration.
// ---------------------------------------------------------------------------
extern "C" void kernel_launch(void* const* d_in, const int* in_sizes, int n_in,
                              void* d_out, int out_size, void* d_ws, size_t ws_size,
                              hipStream_t stream) {
  (void)in_sizes; (void)n_in; (void)out_size; (void)ws_size;
  const int T = 512, Bn = 64, D = 1024, H = 1024, O = 1024;
  const int M = T * Bn;  // 32768

  const float* x        = (const float*)d_in[0];
  const float* paddings = (const float*)d_in[1];
  const float* Wih[2]   = { (const float*)d_in[2], (const float*)d_in[6] };
  const float* Whh[2]   = { (const float*)d_in[3], (const float*)d_in[7] };
  const float* bb [2]   = { (const float*)d_in[4], (const float*)d_in[8] };
  const float* Who[2]   = { (const float*)d_in[5], (const float*)d_in[9] };

  // workspace carve
  char* w = (char*)d_ws;
  float* gx = (float*)w;              w += (size_t)M * 3 * H * 4;
  float* hs = (float*)w;              w += (size_t)(T + 1) * Bn * H * 4;
  unsigned short* hs_bf = (unsigned short*)w; w += (size_t)(T + 1) * Bn * H * 2;
  unsigned short* x_bf  = (unsigned short*)w; w += (size_t)M * D * 2;
  unsigned short* y0_bf = (unsigned short*)w; w += (size_t)M * O * 2;
  unsigned short* Wih_t[2], *Whh_t[2], *Who_t[2];
  for (int l = 0; l < 2; ++l) {
    Wih_t[l] = (unsigned short*)w; w += (size_t)D * 3 * H * 2;
    Whh_t[l] = (unsigned short*)w; w += (size_t)H * 3 * H * 2;
    Who_t[l] = (unsigned short*)w; w += (size_t)H * O * 2;
  }

  // pre-pass: input + weights -> bf16 (weights transposed to [N][K])
  {
    int n4 = M * D / 4;
    cvt_bf16_vec4<<<(n4 + 255) / 256, 256, 0, stream>>>(x, x_bf, n4);
  }
  for (int l = 0; l < 2; ++l) {
    transpose_cvt_bf16<<<dim3(3 * H / 32, D / 32), 256, 0, stream>>>(
        Wih[l], Wih_t[l], D, 3 * H);
    transpose_cvt_bf16<<<dim3(3 * H / 32, H / 32), 256, 0, stream>>>(
        Whh[l], Whh_t[l], H, 3 * H);
    transpose_cvt_bf16<<<dim3(O / 32, H / 32), 256, 0, stream>>>(
        Who[l], Who_t[l], H, O);
  }
  // zero initial hidden state (slot 0, fp32 + bf16)
  zero_u32<<<(Bn * H + 255) / 256, 256, 0, stream>>>((unsigned int*)hs, Bn * H);
  zero_u32<<<(Bn * H / 2 + 255) / 256, 256, 0, stream>>>((unsigned int*)hs_bf,
                                                         Bn * H / 2);

  const unsigned short* layer_in_bf = x_bf;
  for (int l = 0; l < 2; ++l) {
    {  // gx = layer_in @ Wih + b
      dim3 grid(3 * H / GBN, M / GBM);
      gemm_bf16_async<<<grid, 256, 0, stream>>>(layer_in_bf, Wih_t[l], bb[l],
                                                gx, nullptr, M, 3 * H, D);
    }
    for (int t = 0; t < T; ++t) {  // serial recurrence, stream-ordered
      gru_step<<<H / SCT, 256, 0, stream>>>(
          hs + (size_t)t * Bn * H,
          hs_bf + (size_t)t * Bn * H,
          hs + (size_t)(t + 1) * Bn * H,
          hs_bf + (size_t)(t + 1) * Bn * H,
          gx + (size_t)t * Bn * 3 * H,
          paddings + (size_t)t * Bn,
          Whh_t[l]);
    }
    {  // projection: layer0 -> bf16 only (feeds next layer); layer1 -> d_out
      dim3 grid(O / GBN, M / GBM);
      gemm_bf16_async<<<grid, 256, 0, stream>>>(
          hs_bf + (size_t)Bn * H, Who_t[l], nullptr,
          (l == 0) ? nullptr : (float*)d_out,
          (l == 0) ? y0_bf : nullptr,
          M, O, H);
    }
    layer_in_bf = y0_bf;
  }
}
